// HybridAttention_58033598103654
// MI455X (gfx1250) — compile-verified, matrix operand order checked by reading, not run on hardware
//
#include <hip/hip_runtime.h>

typedef __attribute__((ext_vector_type(16))) _Float16 v16h;
typedef __attribute__((ext_vector_type(8)))  _Float16 v8h;
typedef __attribute__((ext_vector_type(8)))  float    v8f;
typedef __attribute__((ext_vector_type(4)))  unsigned v4u;
typedef __attribute__((ext_vector_type(8)))  int      v8i;
typedef __attribute__((ext_vector_type(4)))  int      v4i;

#define WMMA16(a,b,c) __builtin_amdgcn_wmma_f32_16x16x32_f16(false,(a),false,(b),(short)0,(c),false,false)
#define SHUF16(lo,hi) __builtin_shufflevector((lo),(hi),0,1,2,3,4,5,6,7,8,9,10,11,12,13,14,15)

#if defined(__has_builtin)
#  if __has_builtin(__builtin_amdgcn_tensor_load_to_lds)
#    define HAVE_TDM 1
#  endif
#endif
#ifndef HAVE_TDM
#  define HAVE_TDM 0
#endif

constexpr int BB  = 2;
constexpr int T   = 4096;
constexpr int D   = 1024;
constexpr int H   = 16;
constexpr int TC  = 1024;      // T / CF
constexpr int NKP = 1088;      // 1 null + 1024 compressed + 63 zero pad (17*64)
constexpr int N3  = 3 * D;
constexpr int M   = BB * T;    // 8192

// ---------------------------------------------------------------- f32 -> f16
__global__ void cvt_f32_f16(const float* __restrict__ in, _Float16* __restrict__ out, int n) {
  int i = blockIdx.x * blockDim.x + threadIdx.x;
  int stride = gridDim.x * blockDim.x;
  for (; i < n; i += stride) out[i] = (_Float16)in[i];
}

// ------------------------------------------------------- WMMA GEMM (f16->f32)
// mode 0: qkv split epilogue (q->f16, k->f32, v->f32), mode 1: +bias -> f32
__global__ __launch_bounds__(256) void gemm_wmma_f16(
    const _Float16* __restrict__ A, const _Float16* __restrict__ Bm,
    int N, int K, int mode,
    _Float16* __restrict__ qh, float* __restrict__ kf, float* __restrict__ vf,
    const float* __restrict__ bias, float* __restrict__ cf)
{
  __shared__ __align__(16) _Float16 sA[64 * 40];
  __shared__ __align__(16) _Float16 sB[128 * 40];
  const int tid  = threadIdx.x;
  const int lane = tid & 31, w = tid >> 5;
  const int wm = w & 1, wn = w >> 1;
  const int hp = lane >> 4, mr = lane & 15;
  const int m0 = blockIdx.x * 64, n0 = blockIdx.y * 128;

  v8f acc[2][2];
#pragma unroll
  for (int a = 0; a < 2; ++a)
#pragma unroll
    for (int b = 0; b < 2; ++b) acc[a][b] = (v8f){};

  const unsigned long long abase = (unsigned long long)(size_t)A;

  for (int k0 = 0; k0 < K; k0 += 32) {
    __syncthreads();
    {   // stage A tile 64x32 via async copy to LDS (ASYNCcnt path)
      int row = tid >> 2, ks = (tid & 3) * 8;
      unsigned ldsd = (unsigned)(size_t)(sA + row * 40 + ks);
      unsigned voff = (unsigned)(((size_t)(m0 + row) * K + k0 + ks) * 2);
      asm volatile("global_load_async_to_lds_b128 %0, %1, %2"
                   :: "v"(ldsd), "v"(voff), "s"(abase) : "memory");
    }
#pragma unroll
    for (int it = 0; it < 2; ++it) {  // stage B tile transposed: sB[n][k]
      int cid = tid + it * 256;
      int kk = cid >> 4, ns = (cid & 15) * 8;
      v8h val = *(const v8h*)(Bm + (size_t)(k0 + kk) * N + n0 + ns);
#pragma unroll
      for (int e = 0; e < 8; ++e) sB[(ns + e) * 40 + kk] = val[e];
    }
    asm volatile("s_wait_asynccnt 0x0" ::: "memory");
    __syncthreads();

    v16h af[2], bf[2];
#pragma unroll
    for (int mt = 0; mt < 2; ++mt) {
      const _Float16* p = sA + (wm * 32 + mt * 16 + mr) * 40;
      v8h lo = *(const v8h*)(p + hp * 8);
      v8h hi = *(const v8h*)(p + 16 + hp * 8);
      af[mt] = SHUF16(lo, hi);
    }
#pragma unroll
    for (int nt = 0; nt < 2; ++nt) {
      const _Float16* p = sB + (wn * 32 + nt * 16 + mr) * 40;
      v8h lo = *(const v8h*)(p + hp * 8);
      v8h hi = *(const v8h*)(p + 16 + hp * 8);
      bf[nt] = SHUF16(lo, hi);
    }
#pragma unroll
    for (int mt = 0; mt < 2; ++mt)
#pragma unroll
      for (int nt = 0; nt < 2; ++nt)
        acc[mt][nt] = WMMA16(af[mt], bf[nt], acc[mt][nt]);
  }

#pragma unroll
  for (int mt = 0; mt < 2; ++mt)
#pragma unroll
    for (int nt = 0; nt < 2; ++nt) {
      int col = n0 + wn * 32 + nt * 16 + mr;
      size_t row0 = (size_t)m0 + wm * 32 + mt * 16 + hp * 8;
      if (mode == 0) {
        if (col < D) {
#pragma unroll
          for (int r = 0; r < 8; ++r) qh[(row0 + r) * D + col] = (_Float16)acc[mt][nt][r];
        } else if (col < 2 * D) {
#pragma unroll
          for (int r = 0; r < 8; ++r) kf[(row0 + r) * D + (col - D)] = acc[mt][nt][r];
        } else {
#pragma unroll
          for (int r = 0; r < 8; ++r) vf[(row0 + r) * D + (col - 2 * D)] = acc[mt][nt][r];
        }
      } else {
        float bv = bias[col];
#pragma unroll
        for (int r = 0; r < 8; ++r) cf[(row0 + r) * N + col] = acc[mt][nt][r] + bv;
      }
    }
}

// -------------------------------------------- null rows + zero padding of k/v
__global__ void prep_kv(const float* __restrict__ nk, const float* __restrict__ nv,
                        _Float16* __restrict__ kc, _Float16* __restrict__ vc) {
  int i = blockIdx.x * 256 + threadIdx.x;          // BB * 64 * D
  if (i >= BB * 64 * D) return;
  int d = i & (D - 1), r = (i >> 10) & 63, b = i >> 16;
  int row = (r == 0) ? 0 : (TC + r);               // row 0 = null, 1025..1087 = pad
  float fk = (r == 0) ? nk[d] : 0.f;
  float fv = (r == 0) ? nv[d] : 0.f;
  size_t off = ((size_t)b * NKP + row) * D + d;
  kc[off] = (_Float16)fk;
  vc[off] = (_Float16)fv;
}

// --------------------------------- grouped conv1d(k=4, stride=4, groups=16)
__global__ __launch_bounds__(64) void compress_kv(
    const float* __restrict__ kin, const float* __restrict__ vin,
    const float* __restrict__ wgt, const float* __restrict__ bias,
    _Float16* __restrict__ kc, _Float16* __restrict__ vc)
{
  __shared__ float xk[256], xv[256];
  int bid = blockIdx.x;
  int g = bid & (H - 1), tc = (bid >> 4) & (TC - 1), b = bid >> 14;
  int tid = threadIdx.x;
#pragma unroll
  for (int j = 0; j < 4; ++j) {
    size_t src = ((size_t)(b * T + tc * 4 + j)) * D + g * 64 + tid;
    xk[j * 64 + tid] = kin[src];
    xv[j * 64 + tid] = vin[src];
  }
  __syncthreads();
  int oc = g * 64 + tid;
  float ak = bias[oc], av = bias[oc];
  const float* wp = wgt + (size_t)oc * 256;
#pragma unroll 4
  for (int ic = 0; ic < 64; ++ic)
#pragma unroll
    for (int j = 0; j < 4; ++j) {
      float wv = wp[ic * 4 + j];
      ak = fmaf(xk[j * 64 + ic], wv, ak);
      av = fmaf(xv[j * 64 + ic], wv, av);
    }
  size_t dst = ((size_t)b * NKP + 1 + tc) * D + oc;
  kc[dst] = (_Float16)ak;
  vc[dst] = (_Float16)av;
}

// -------------------------------------- flash attention, block-causal mask
__global__ __launch_bounds__(128) void attn_kernel(
    const _Float16* __restrict__ Q, const _Float16* __restrict__ Kc,
    const _Float16* __restrict__ Vc, _Float16* __restrict__ Oout)
{
  __shared__ __align__(16) _Float16 sK[64 * 72];        // [key][dh], 16B row pad
  __shared__ __align__(16) _Float16 sV[64 * 72];        // [dh][key] (transposed)
  __shared__ __align__(16) _Float16 sP[4 * 16 * 72];    // per-wave P tile
  const int tid = threadIdx.x, lane = tid & 31, w = tid >> 5;
  const int hp = lane >> 4, mr = lane & 15;
  const int q0 = blockIdx.x * 64;
  const int bh = blockIdx.y, b = bh >> 4, h = bh & 15;
  const int q0w = q0 + w * 16;

  // Q fragments (16 queries x 64 dh) straight from global
  const _Float16* qp = Q + ((size_t)(b * T) + q0w + mr) * D + h * 64;
  v16h aq0, aq1;
  { v8h lo = *(const v8h*)(qp + hp * 8);      v8h hi = *(const v8h*)(qp + 16 + hp * 8); aq0 = SHUF16(lo, hi); }
  { v8h lo = *(const v8h*)(qp + 32 + hp * 8); v8h hi = *(const v8h*)(qp + 48 + hp * 8); aq1 = SHUF16(lo, hi); }

  v8f Oacc[4]; v8f mst, lst;
#pragma unroll
  for (int nt = 0; nt < 4; ++nt) Oacc[nt] = (v8f){};
#pragma unroll
  for (int r = 0; r < 8; ++r) { mst[r] = -1e30f; lst[r] = 0.f; }

  const int jmax = min(TC, q0 / 4 + 16);
  const int nblk = (jmax + 64) >> 6;
  _Float16* pbase = sP + w * 16 * 72;

  for (int blk = 0; blk < nblk; ++blk) {
    const int kb = blk * 64;
    __syncthreads();

#if HAVE_TDM
    if (w == 0) {
      // Tensor Data Mover: 2D tile 64 keys x 64 dh, f16, row stride D,
      // LDS padding 16B per 128B row -> matches sK stride of 72 halves.
      unsigned long long ga =
          (unsigned long long)(size_t)(Kc + ((size_t)b * NKP + kb) * D + h * 64);
      v4u g0;
      g0[0] = 1u;                                   // count = 1 valid descriptor
      g0[1] = (unsigned)(size_t)sK;                 // lds_addr
      g0[2] = (unsigned)ga;                         // global_addr[31:0]
      g0[3] = (unsigned)(ga >> 32) | 0x80000000u;   // global_addr[56:32] | type=2
      v8i g1 = (v8i){};
      g1[0] = 0x07110000;        // data_size=2B | pad_enable | pad_interval=32dw | pad_amount=4dw
      g1[1] = (int)(64u << 16);  // tensor_dim0 = 64
      g1[2] = (int)(64u << 16);  // tensor_dim1 = 64
      g1[3] = (int)(64u << 16);  // tile_dim0 = 64
      g1[4] = 64;                // tile_dim1 = 64, tile_dim2 = 0
      g1[5] = 1024;              // tensor_dim0_stride = D
      v4i gz = (v4i){};
#if __clang_major__ >= 23
      __builtin_amdgcn_tensor_load_to_lds(g0, g1, gz, gz, (v8i){}, 0);
#else
      __builtin_amdgcn_tensor_load_to_lds(g0, g1, gz, gz, 0);
#endif
    }
#endif

#pragma unroll
    for (int it = 0; it < 4; ++it) {   // stage V (transposed); K too if no TDM
      int cid = tid + it * 128;
      int key = cid >> 3, ks = (cid & 7) * 8;
      size_t srow = ((size_t)b * NKP + kb + key) * D + h * 64 + ks;
#if !HAVE_TDM
      *(v8h*)(sK + key * 72 + ks) = *(const v8h*)(Kc + srow);
#endif
      v8h vv = *(const v8h*)(Vc + srow);
#pragma unroll
      for (int e = 0; e < 8; ++e) sV[(ks + e) * 72 + key] = vv[e];
    }
#if HAVE_TDM
    if (w == 0) __builtin_amdgcn_s_wait_tensorcnt(0);
#endif
    __syncthreads();

    // S = Q K^T
    v8f sc[4];
#pragma unroll
    for (int kt = 0; kt < 4; ++kt) {
      const _Float16* kp = sK + (kt * 16 + mr) * 72;
      v16h blo, bhi;
      { v8h lo = *(const v8h*)(kp + hp * 8);      v8h hi = *(const v8h*)(kp + 16 + hp * 8); blo = SHUF16(lo, hi); }
      { v8h lo = *(const v8h*)(kp + 32 + hp * 8); v8h hi = *(const v8h*)(kp + 48 + hp * 8); bhi = SHUF16(lo, hi); }
      v8f z = (v8f){};
      z = WMMA16(aq0, blo, z);
      z = WMMA16(aq1, bhi, z);
      sc[kt] = z;
    }
    // scale + block-causal mask (allowed: j==0 || i >= 4j-1)
#pragma unroll
    for (int kt = 0; kt < 4; ++kt) {
      int j = kb + kt * 16 + mr;
#pragma unroll
      for (int r = 0; r < 8; ++r) {
        int i = q0w + hp * 8 + r;
        float s = sc[kt][r] * 0.03125f;   // 1/sqrt(1024)
        bool ok = (j == 0) || ((j <= TC) && (i >= 4 * j - 1));
        sc[kt][r] = ok ? s : -1e30f;
      }
    }
    // online softmax (row stats componentwise, reduced across 16-lane half)
    v8f bm;
#pragma unroll
    for (int r = 0; r < 8; ++r)
      bm[r] = fmaxf(fmaxf(sc[0][r], sc[1][r]), fmaxf(sc[2][r], sc[3][r]));
#pragma unroll
    for (int dlt = 1; dlt <= 8; dlt <<= 1)
#pragma unroll
      for (int r = 0; r < 8; ++r) bm[r] = fmaxf(bm[r], __shfl_xor(bm[r], dlt, 32));
    v8f mnew, alpha, rs = (v8f){};
#pragma unroll
    for (int r = 0; r < 8; ++r) {
      mnew[r] = fmaxf(mst[r], bm[r]);
      alpha[r] = __expf(mst[r] - mnew[r]);
    }
#pragma unroll
    for (int kt = 0; kt < 4; ++kt)
#pragma unroll
      for (int r = 0; r < 8; ++r) {
        float p = __expf(sc[kt][r] - mnew[r]);
        sc[kt][r] = p;
        rs[r] += p;
      }
#pragma unroll
    for (int dlt = 1; dlt <= 8; dlt <<= 1)
#pragma unroll
      for (int r = 0; r < 8; ++r) rs[r] += __shfl_xor(rs[r], dlt, 32);
#pragma unroll
    for (int r = 0; r < 8; ++r) {
      lst[r] = lst[r] * alpha[r] + rs[r];
      mst[r] = mnew[r];
    }
#pragma unroll
    for (int nt = 0; nt < 4; ++nt)
#pragma unroll
      for (int r = 0; r < 8; ++r) Oacc[nt][r] *= alpha[r];

    // C-layout -> A-fragment via per-wave LDS tile
#pragma unroll
    for (int kt = 0; kt < 4; ++kt)
#pragma unroll
      for (int r = 0; r < 8; ++r)
        pbase[(hp * 8 + r) * 72 + kt * 16 + mr] = (_Float16)sc[kt][r];
    asm volatile("s_wait_dscnt 0" ::: "memory");   // same-wave LDS write->read
    const _Float16* pp = pbase + mr * 72;
    v16h ap0, ap1;
    { v8h lo = *(const v8h*)(pp + hp * 8);      v8h hi = *(const v8h*)(pp + 16 + hp * 8); ap0 = SHUF16(lo, hi); }
    { v8h lo = *(const v8h*)(pp + 32 + hp * 8); v8h hi = *(const v8h*)(pp + 48 + hp * 8); ap1 = SHUF16(lo, hi); }

    // O += P V
#pragma unroll
    for (int nt = 0; nt < 4; ++nt) {
      const _Float16* vp = sV + (nt * 16 + mr) * 72;
      v16h bl, bh2;
      { v8h lo = *(const v8h*)(vp + hp * 8);      v8h hi = *(const v8h*)(vp + 16 + hp * 8); bl  = SHUF16(lo, hi); }
      { v8h lo = *(const v8h*)(vp + 32 + hp * 8); v8h hi = *(const v8h*)(vp + 48 + hp * 8); bh2 = SHUF16(lo, hi); }
      Oacc[nt] = WMMA16(ap0, bl, Oacc[nt]);
      Oacc[nt] = WMMA16(ap1, bh2, Oacc[nt]);
    }
  }

#pragma unroll
  for (int nt = 0; nt < 4; ++nt)
#pragma unroll
    for (int r = 0; r < 8; ++r) {
      float v = Oacc[nt][r] / lst[r];
      Oout[((size_t)(b * T) + q0w + hp * 8 + r) * D + h * 64 + nt * 16 + mr] = (_Float16)v;
    }
}

// ------------------------------------------ 1024-pt FFT of real rows (axis -1)
__global__ __launch_bounds__(256) void fft1024_rows(const float* __restrict__ in,
                                                    float2* __restrict__ out) {
  __shared__ float re[1024], im[1024];
  const int row = blockIdx.x, tid = threadIdx.x;
  const float* src = in + (size_t)row * 1024;
#pragma unroll
  for (int s = 0; s < 4; ++s) {
    int i = tid + s * 256;
    int j = __brev((unsigned)i) >> 22;
    re[i] = src[j];
    im[i] = 0.f;
  }
  __syncthreads();
  for (int len = 2; len <= 1024; len <<= 1) {
    int hl = len >> 1;
#pragma unroll
    for (int s = 0; s < 2; ++s) {
      int t = tid + s * 256;
      int p = t & (hl - 1);
      int i0 = ((t - p) << 1) + p;
      int i1 = i0 + hl;
      float ang = -6.283185307179586f * (float)p / (float)len;
      float sn, cs;
      __sincosf(ang, &sn, &cs);
      float xr = re[i1], xi = im[i1];
      float tr = xr * cs - xi * sn;
      float ti = xr * sn + xi * cs;
      re[i1] = re[i0] - tr; im[i1] = im[i0] - ti;
      re[i0] += tr;         im[i0] += ti;
    }
    __syncthreads();
  }
  float2* dst = out + (size_t)row * 1024;
#pragma unroll
  for (int s = 0; s < 4; ++s) {
    int i = tid + s * 256;
    dst[i] = make_float2(re[i], im[i]);
  }
}

// ------------------------------- 4096-pt complex FFT along t, keep real part
__global__ __launch_bounds__(512) void fft4096_cols(const float2* __restrict__ in,
                                                    float* __restrict__ out) {
  __shared__ float re[4096], im[4096];
  const int cid = blockIdx.x, tid = threadIdx.x;
  const int b = cid >> 10, fd = cid & 1023;
  const float2* src = in + (size_t)b * T * 1024 + fd;
#pragma unroll
  for (int s = 0; s < 8; ++s) {
    int i = tid + s * 512;
    int j = __brev((unsigned)i) >> 20;
    float2 v = src[(size_t)j * 1024];
    re[i] = v.x; im[i] = v.y;
  }
  __syncthreads();
  for (int len = 2; len <= 4096; len <<= 1) {
    int hl = len >> 1;
#pragma unroll
    for (int s = 0; s < 4; ++s) {
      int t = tid + s * 512;
      int p = t & (hl - 1);
      int i0 = ((t - p) << 1) + p;
      int i1 = i0 + hl;
      float ang = -6.283185307179586f * (float)p / (float)len;
      float sn, cs;
      __sincosf(ang, &sn, &cs);
      float xr = re[i1], xi = im[i1];
      float tr = xr * cs - xi * sn;
      float ti = xr * sn + xi * cs;
      re[i1] = re[i0] - tr; im[i1] = im[i0] - ti;
      re[i0] += tr;         im[i0] += ti;
    }
    __syncthreads();
  }
  float* dst = out + (size_t)b * T * 1024 + fd;
#pragma unroll
  for (int s = 0; s < 8; ++s) {
    int i = tid + s * 512;
    dst[(size_t)i * 1024] = re[i];
  }
}

// ----------------------------------------------------------------- launcher
extern "C" void kernel_launch(void* const* d_in, const int* in_sizes, int n_in,
                              void* d_out, int out_size, void* d_ws, size_t ws_size,
                              hipStream_t stream) {
  const float* x      = (const float*)d_in[0];
  const float* Wqkv   = (const float*)d_in[1];
  const float* conv_w = (const float*)d_in[2];
  const float* conv_b = (const float*)d_in[3];
  const float* null_k = (const float*)d_in[4];
  const float* null_v = (const float*)d_in[5];
  const float* W_out  = (const float*)d_in[6];
  const float* b_out  = (const float*)d_in[7];
  (void)in_sizes; (void)n_in; (void)out_size; (void)ws_size;

  char* ws = (char*)d_ws;
  size_t off = 0;
  auto alloc = [&](size_t bytes) {
    void* p = ws + off;
    off += (bytes + 255) & ~(size_t)255;
    return p;
  };
  _Float16* x_h    = (_Float16*)alloc((size_t)M * D * 2);
  _Float16* wqkv_h = (_Float16*)alloc((size_t)D * N3 * 2);
  _Float16* wout_h = (_Float16*)alloc((size_t)D * D * 2);
  _Float16* q_h    = (_Float16*)alloc((size_t)M * D * 2);
  _Float16* kc_h   = (_Float16*)alloc((size_t)BB * NKP * D * 2);
  _Float16* vc_h   = (_Float16*)alloc((size_t)BB * NKP * D * 2);
  _Float16* ao_h   = (_Float16*)alloc((size_t)M * D * 2);
  float*    proj   = (float*)alloc((size_t)M * D * 4);
  float*    k_f    = (float*)alloc((size_t)M * D * 4);
  float*    v_f    = (float*)alloc((size_t)M * D * 4);
  float2*   fftc   = (float2*)(void*)k_f;   // overlay k_f+v_f (compress done before FFT)

  cvt_f32_f16<<<2048, 256, 0, stream>>>(x, x_h, M * D);
  cvt_f32_f16<<<2048, 256, 0, stream>>>(Wqkv, wqkv_h, D * N3);
  cvt_f32_f16<<<1024, 256, 0, stream>>>(W_out, wout_h, D * D);

  gemm_wmma_f16<<<dim3(M / 64, N3 / 128), 256, 0, stream>>>(
      x_h, wqkv_h, N3, D, 0, q_h, k_f, v_f, nullptr, nullptr);

  prep_kv<<<(BB * 64 * D) / 256, 256, 0, stream>>>(null_k, null_v, kc_h, vc_h);
  compress_kv<<<BB * TC * H, 64, 0, stream>>>(k_f, v_f, conv_w, conv_b, kc_h, vc_h);

  attn_kernel<<<dim3(T / 64, BB * H), 128, 0, stream>>>(q_h, kc_h, vc_h, ao_h);

  gemm_wmma_f16<<<dim3(M / 64, D / 128), 256, 0, stream>>>(
      ao_h, wout_h, D, D, 1, nullptr, nullptr, nullptr, b_out, proj);

  fft1024_rows<<<M, 256, 0, stream>>>(proj, fftc);
  fft4096_cols<<<BB * D, 512, 0, stream>>>(fftc, (float*)d_out);
}
